// EncoderBlock_80161269613130
// MI455X (gfx1250) — matrix operand order repaired and checked by reading, not compiled, not run on hardware
//
#include <hip/hip_runtime.h>

// ---------------- problem constants ----------------
#define BATCH  4
#define SEQ    2048
#define DMODEL 512
#define NHEAD  8
#define DKH    64
#define DFFN   2048
#define BSROWS (BATCH * SEQ)          // 8192

typedef __bf16 bf16_t;
typedef __attribute__((ext_vector_type(16))) __bf16 v16bf;
typedef __attribute__((ext_vector_type(8)))  __bf16 v8bf;
typedef __attribute__((ext_vector_type(4)))  __bf16 v4bf;
typedef __attribute__((ext_vector_type(8)))  float  v8f;
typedef __attribute__((ext_vector_type(4)))  int    v4i;

// ---------------- helpers ----------------
static __device__ __forceinline__ bf16_t f2bf(float f) {
  unsigned u = __builtin_bit_cast(unsigned, f);
  u += 0x7fffu + ((u >> 16) & 1u);               // round-to-nearest-even
  unsigned short h = (unsigned short)(u >> 16);
  return __builtin_bit_cast(bf16_t, h);
}

static __device__ __forceinline__ v8f wmma_bf16(v16bf a, v16bf b, v8f c) {
  // D(f32 16x16) = A(16x32 bf16) * B(32x16 bf16) + C
  return __builtin_amdgcn_wmma_f32_16x16x32_bf16(
      /*neg_a=*/false, a, /*neg_b=*/false, b,
      /*c_mod=*/(short)0, c, /*reuse_a=*/false, /*reuse_b=*/false);
}

// ---- async global -> LDS copy (gfx1250 ASYNCcnt path) ----
static __device__ __forceinline__ void async_copy_b128(const bf16_t* src,
                                                       bf16_t* dst_lds) {
#if __has_builtin(__builtin_amdgcn_global_load_async_to_lds_b128)
  __builtin_amdgcn_global_load_async_to_lds_b128((v4i*)src, (v4i*)dst_lds,
                                                 0, 0);
#else
  asm volatile("global_load_async_to_lds_b128 %0, %1, off"
               :: "v"((unsigned)(size_t)dst_lds), "v"(src)
               : "memory");
#endif
}
static __device__ __forceinline__ void wait_async_le1() {
  asm volatile("s_wait_asynccnt 1" ::: "memory");
}
static __device__ __forceinline__ void wait_async_le0() {
  asm volatile("s_wait_asynccnt 0" ::: "memory");
}

// A fragment: 16(M) x 32(K). Source row-major [*, lda]; rows row0..row0+15,
// cols k..k+31. ISA layout: lanes 0-15 hold K 0-7 & 16-23, lanes 16-31 hold
// K 8-15 & 24-31 (per 16-bit A-matrix table).
static __device__ __forceinline__ v16bf load_frag_a(const bf16_t* p, int lda,
                                                    int row0, int k) {
  const int lane = threadIdx.x & 31;
  const int g = lane >> 4, m = lane & 15;
  const bf16_t* rp = p + (size_t)(row0 + m) * lda + k + g * 8;
  v8bf lo = *(const v8bf*)rp;          // global_load_b128
  v8bf hi = *(const v8bf*)(rp + 16);   // global_load_b128
  v16bf r;
#pragma unroll
  for (int i = 0; i < 8; ++i) { r[i] = lo[i]; r[i + 8] = hi[i]; }
  return r;
}

// Same A-fragment pattern but from an LDS-resident 16x32 tile.
static __device__ __forceinline__ v16bf lds_frag_a(const bf16_t* lds) {
  const int lane = threadIdx.x & 31;
  const int g = lane >> 4, m = lane & 15;
  const bf16_t* rp = lds + m * 32 + g * 8;
  v8bf lo = *(const v8bf*)rp;          // ds_load_b128
  v8bf hi = *(const v8bf*)(rp + 16);
  v16bf r;
#pragma unroll
  for (int i = 0; i < 8; ++i) { r[i] = lo[i]; r[i + 8] = hi[i]; }
  return r;
}

// B fragment: 32(K) x 16(N) with B[kk][n] = src[n0+n][k+kk]; src row-major
// [*, ldb]. ISA layout: lanes 0-15 hold K 0-7 & 16-23 of column n, lanes 16-31
// hold K 8-15 & 24-31 of column n-16 (matching the A fragment).
static __device__ __forceinline__ v16bf load_frag_b(const bf16_t* p, int ldb,
                                                    int n0, int k) {
  const int lane = threadIdx.x & 31;
  const int g = lane >> 4, n = lane & 15;
  const bf16_t* rp = p + (size_t)(n0 + n) * ldb + k + g * 8;
  v8bf lo = *(const v8bf*)rp;          // global_load_b128
  v8bf hi = *(const v8bf*)(rp + 16);
  v16bf r;
#pragma unroll
  for (int i = 0; i < 8; ++i) { r[i] = lo[i]; r[i + 8] = hi[i]; }
  return r;
}

// Same B-fragment pattern from an LDS-resident 64x32 tile (row-major, 32/row).
static __device__ __forceinline__ v16bf lds_frag_b(const bf16_t* lds, int n0) {
  const int lane = threadIdx.x & 31;
  const int g = lane >> 4, n = lane & 15;
  const bf16_t* rp = lds + (n0 + n) * 32 + g * 8;
  v8bf lo = *(const v8bf*)rp;          // ds_load_b128
  v8bf hi = *(const v8bf*)(rp + 16);
  v16bf r;
#pragma unroll
  for (int i = 0; i < 8; ++i) { r[i] = lo[i]; r[i + 8] = hi[i]; }
  return r;
}

// Stage a shared 64(rows) x 32(k) bf16 weight tile into LDS: 256 threads,
// one async b128 per thread.
static __device__ __forceinline__ void async_stage_b64(const bf16_t* w, int ldb,
                                                       int n0, int kk,
                                                       bf16_t* lds) {
  const int t = threadIdx.x;           // 0..255
  const int row = t >> 2;              // 0..63
  const int c4 = (t & 3) * 8;          // 0,8,16,24
  async_copy_b128(w + (size_t)(n0 + row) * ldb + kk + c4, lds + row * 32 + c4);
}

// Stage a shared 16(rows) x 32(k) bf16 activation tile into LDS (64 threads).
static __device__ __forceinline__ void async_stage_a16(const bf16_t* a, int lda,
                                                       int row0, int kk,
                                                       bf16_t* lds) {
  const int t = threadIdx.x;
  if (t < 64) {
    const int row = t >> 2;            // 0..15
    const int c4 = (t & 3) * 8;
    async_copy_b128(a + (size_t)(row0 + row) * lda + kk + c4,
                    lds + row * 32 + c4);
  }
}

// ---------------- fp32 -> bf16 conversion ----------------
__global__ void cvt_f32_bf16(const float* __restrict__ in,
                             bf16_t* __restrict__ out, int n4) {
  int i = blockIdx.x * 256 + threadIdx.x;
  if (i < n4) {
    float4 f = ((const float4*)in)[i];
    v4bf o; o[0] = f2bf(f.x); o[1] = f2bf(f.y); o[2] = f2bf(f.z); o[3] = f2bf(f.w);
    ((v4bf*)out)[i] = o;
  }
}

// ---------------- fused QKV projection ----------------
// out = x @ W^T + b.  grid: (BS/128, D/64, 3). block: 256 (8 waves x 16 rows).
// Shared B tile double-buffered through LDS via async copies.
// z==0 -> Q [B,H,S,DK] bf16 ; z==1 -> K [B,H,S,DK] ; z==2 -> V^T [B,H,DK,S].
__global__ void gemm_qkv(const bf16_t* __restrict__ xb,
                         const bf16_t* __restrict__ wqb,
                         const bf16_t* __restrict__ wkb,
                         const bf16_t* __restrict__ wvb,
                         const float* __restrict__ bq,
                         const float* __restrict__ bk,
                         const float* __restrict__ bv,
                         bf16_t* __restrict__ q, bf16_t* __restrict__ k,
                         bf16_t* __restrict__ vT) {
  __shared__ __align__(16) bf16_t lds_b[2][64 * 32];   // 2 x 4 KB
  const int wave = threadIdx.x >> 5;
  const int row0 = blockIdx.x * 128 + wave * 16;
  const int n0   = blockIdx.y * 64;
  const int z    = blockIdx.z;
  const bf16_t* w   = (z == 0) ? wqb : (z == 1) ? wkb : wvb;
  const float* bias = (z == 0) ? bq  : (z == 1) ? bk  : bv;

  v8f acc[4];
#pragma unroll
  for (int t = 0; t < 4; ++t)
#pragma unroll
    for (int r = 0; r < 8; ++r) acc[t][r] = 0.f;

  const int steps = DMODEL / 32;       // 16
  async_stage_b64(w, DMODEL, n0, 0, lds_b[0]);
  for (int i = 0; i < steps; ++i) {
    if (i + 1 < steps) {
      async_stage_b64(w, DMODEL, n0, (i + 1) * 32, lds_b[(i + 1) & 1]);
      wait_async_le1();                // current buffer's copy retired
    } else {
      wait_async_le0();
    }
    __syncthreads();                   // everyone's LDS writes visible
    v16bf a = load_frag_a(xb, DMODEL, row0, i * 32);
    const bf16_t* bt = lds_b[i & 1];
#pragma unroll
    for (int t = 0; t < 4; ++t)
      acc[t] = wmma_bf16(a, lds_frag_b(bt, t * 16), acc[t]);
    __syncthreads();                   // reads done before buffer reuse
  }

  const int lane = threadIdx.x & 31;
  const int g = lane >> 4, ln = lane & 15;
#pragma unroll
  for (int t = 0; t < 4; ++t) {
    const int col = n0 + t * 16 + ln;
    const float bb = bias[col];
    const int h = col >> 6, d = col & 63;
#pragma unroll
    for (int r = 0; r < 8; ++r) {
      const int row = row0 + r + g * 8;
      const int b = row >> 11, s = row & (SEQ - 1);
      const bf16_t o = f2bf(acc[t][r] + bb);
      if (z == 2)
        vT[(((size_t)b * NHEAD + h) * DKH + d) * SEQ + s] = o;
      else if (z == 0)
        q[(((size_t)b * NHEAD + h) * SEQ + s) * DKH + d] = o;
      else
        k[(((size_t)b * NHEAD + h) * SEQ + s) * DKH + d] = o;
    }
  }
}

// ---------------- flash attention ----------------
// grid: (B*H, S/16/4), block: 128 (4 waves, 1 q-tile of 16 rows each).
__global__ void flash_attn(const bf16_t* __restrict__ q,
                           const bf16_t* __restrict__ kmat,
                           const bf16_t* __restrict__ vT,
                           const unsigned char* __restrict__ mask,
                           bf16_t* __restrict__ o) {
  __shared__ __align__(16) bf16_t lds_p[4][16 * 32];   // P staging per wave
  const int wave = threadIdx.x >> 5;
  const int lane = threadIdx.x & 31;
  const int g = lane >> 4, ln = lane & 15;
  const int bh = blockIdx.x;          // 0..31
  const int b  = bh >> 3;             // H = 8
  const int h  = bh & 7;
  const int s0 = (blockIdx.y * 4 + wave) * 16;

  const bf16_t* qb = q    + (size_t)bh * SEQ * DKH;
  const bf16_t* kb = kmat + (size_t)bh * SEQ * DKH;
  const bf16_t* vb = vT   + (size_t)bh * DKH * SEQ;
  const unsigned char* mb = mask + (size_t)b * SEQ;

  // Q fragments (16 x 64), kept resident
  v16bf qa0 = load_frag_a(qb, DKH, s0, 0);
  v16bf qa1 = load_frag_a(qb, DKH, s0, 32);

  v8f acc[4];
  float mrow[8], lrow[8];
#pragma unroll
  for (int t = 0; t < 4; ++t)
#pragma unroll
    for (int r = 0; r < 8; ++r) acc[t][r] = 0.f;
#pragma unroll
  for (int r = 0; r < 8; ++r) { mrow[r] = -1e30f; lrow[r] = 0.f; }

  const float scale = 0.125f;  // 1/sqrt(64)

  for (int j = 0; j < SEQ; j += 32) {
    // scores: 16 q x 32 keys  (two 16x16 tiles, dk=64 => 2 WMMA each)
    v8f sc0, sc1;
#pragma unroll
    for (int r = 0; r < 8; ++r) { sc0[r] = 0.f; sc1[r] = 0.f; }
    sc0 = wmma_bf16(qa0, load_frag_b(kb, DKH, j, 0),       sc0);
    sc0 = wmma_bf16(qa1, load_frag_b(kb, DKH, j, 32),      sc0);
    sc1 = wmma_bf16(qa0, load_frag_b(kb, DKH, j + 16, 0),  sc1);
    sc1 = wmma_bf16(qa1, load_frag_b(kb, DKH, j + 16, 32), sc1);

    const bool mk0 = mb[j + ln] != 0;        // mask depends only on key index
    const bool mk1 = mb[j + 16 + ln] != 0;
#pragma unroll
    for (int r = 0; r < 8; ++r) {
      sc0[r] = mk0 ? -1.0e9f : sc0[r] * scale;
      sc1[r] = mk1 ? -1.0e9f : sc1[r] * scale;
    }

    // online softmax: row-wise over the 16-lane group
    float mnew[8];
#pragma unroll
    for (int r = 0; r < 8; ++r) mnew[r] = fmaxf(mrow[r], fmaxf(sc0[r], sc1[r]));
#pragma unroll
    for (int off = 1; off <= 8; off <<= 1)
#pragma unroll
      for (int r = 0; r < 8; ++r)
        mnew[r] = fmaxf(mnew[r], __shfl_xor(mnew[r], off, 32));

    float p0[8], p1[8], alpha[8], lsum[8];
#pragma unroll
    for (int r = 0; r < 8; ++r) {
      alpha[r] = __expf(mrow[r] - mnew[r]);
      p0[r]    = __expf(sc0[r] - mnew[r]);
      p1[r]    = __expf(sc1[r] - mnew[r]);
      mrow[r]  = mnew[r];
      lsum[r]  = p0[r] + p1[r];
    }
#pragma unroll
    for (int off = 1; off <= 8; off <<= 1)
#pragma unroll
      for (int r = 0; r < 8; ++r) lsum[r] += __shfl_xor(lsum[r], off, 32);
#pragma unroll
    for (int r = 0; r < 8; ++r) lrow[r] = lrow[r] * alpha[r] + lsum[r];
#pragma unroll
    for (int t = 0; t < 4; ++t)
#pragma unroll
      for (int r = 0; r < 8; ++r) acc[t][r] *= alpha[r];

    // stage P (16x32) as bf16 in LDS, reload in A-fragment layout
#pragma unroll
    for (int r = 0; r < 8; ++r) {
      const int m = r + g * 8;
      lds_p[wave][m * 32 + ln]      = f2bf(p0[r]);
      lds_p[wave][m * 32 + 16 + ln] = f2bf(p1[r]);
    }
    asm volatile("s_wait_dscnt 0" ::: "memory");
    v8bf plo = *(const v8bf*)&lds_p[wave][ln * 32 + g * 8];
    v8bf phi = *(const v8bf*)&lds_p[wave][ln * 32 + 16 + g * 8];
    v16bf pf;
#pragma unroll
    for (int i = 0; i < 8; ++i) { pf[i] = plo[i]; pf[i + 8] = phi[i]; }

    // acc += P(16x32) * V(32keys x 64)  -- V stored transposed [DK, S]
#pragma unroll
    for (int t = 0; t < 4; ++t)
      acc[t] = wmma_bf16(pf, load_frag_b(vb, SEQ, t * 16, j), acc[t]);
  }

  float inv[8];
#pragma unroll
  for (int r = 0; r < 8; ++r) inv[r] = (lrow[r] > 0.f) ? 1.f / lrow[r] : 0.f;

  // write o as bf16, layout [B, S, H*DK] = [BS, D]
  const size_t base = ((size_t)b * SEQ + s0) * DMODEL + h * DKH;
#pragma unroll
  for (int t = 0; t < 4; ++t)
#pragma unroll
    for (int r = 0; r < 8; ++r)
      o[base + (size_t)(r + g * 8) * DMODEL + t * 16 + ln] =
          f2bf(acc[t][r] * inv[r]);
}

// ---------------- GEMM + bias + ReLU (FFN layer 1) ----------------
// grid: (BS/128, DFF/64), block 256. h[row, col] = relu(y @ w1^T + b1), bf16.
// Shared B tile double-buffered through LDS via async copies.
__global__ void gemm_ffn1(const bf16_t* __restrict__ yb,
                          const bf16_t* __restrict__ w1b,
                          const float* __restrict__ b1,
                          bf16_t* __restrict__ hbuf) {
  __shared__ __align__(16) bf16_t lds_b[2][64 * 32];
  const int wave = threadIdx.x >> 5;
  const int row0 = blockIdx.x * 128 + wave * 16;
  const int n0   = blockIdx.y * 64;

  v8f acc[4];
#pragma unroll
  for (int t = 0; t < 4; ++t)
#pragma unroll
    for (int r = 0; r < 8; ++r) acc[t][r] = 0.f;

  const int steps = DMODEL / 32;
  async_stage_b64(w1b, DMODEL, n0, 0, lds_b[0]);
  for (int i = 0; i < steps; ++i) {
    if (i + 1 < steps) {
      async_stage_b64(w1b, DMODEL, n0, (i + 1) * 32, lds_b[(i + 1) & 1]);
      wait_async_le1();
    } else {
      wait_async_le0();
    }
    __syncthreads();
    v16bf a = load_frag_a(yb, DMODEL, row0, i * 32);
    const bf16_t* bt = lds_b[i & 1];
#pragma unroll
    for (int t = 0; t < 4; ++t)
      acc[t] = wmma_bf16(a, lds_frag_b(bt, t * 16), acc[t]);
    __syncthreads();
  }

  const int lane = threadIdx.x & 31;
  const int g = lane >> 4, ln = lane & 15;
#pragma unroll
  for (int t = 0; t < 4; ++t) {
    const int col = n0 + t * 16 + ln;
    const float bb = b1[col];
#pragma unroll
    for (int r = 0; r < 8; ++r) {
      float v = acc[t][r] + bb;
      v = v > 0.f ? v : 0.f;                    // ReLU
      hbuf[(size_t)(row0 + r + g * 8) * DFFN + col] = f2bf(v);
    }
  }
}

// ---------------- GEMM + bias + LayerNorm + residual ----------------
// One block owns a full 16-row x 512-col tile (8 waves x 64 cols), so the
// LayerNorm (unbiased var, ddof=1) reduces fully inside the block. The shared
// 16x32 A tile is double-buffered through LDS via async copies.
// out_f32 = res + gamma*(t-mean)/sqrt(var+eps) + beta ; out_bf optional copy.
__global__ void gemm_ln_res(const bf16_t* __restrict__ A,
                            const bf16_t* __restrict__ W,
                            const float* __restrict__ bias,
                            const float* __restrict__ res,
                            const float* __restrict__ gamma,
                            const float* __restrict__ beta,
                            float* __restrict__ out_f32,
                            bf16_t* __restrict__ out_bf, int K) {
  __shared__ __align__(16) bf16_t lds_a[2][16 * 32];   // 2 x 1 KB
  __shared__ float tile[16 * DMODEL];                  // 32 KB
  const int wave = threadIdx.x >> 5;
  const int lane = threadIdx.x & 31;
  const int g = lane >> 4, ln = lane & 15;
  const int row0 = blockIdx.x * 16;
  const int n0   = wave * 64;

  v8f acc[4];
#pragma unroll
  for (int t = 0; t < 4; ++t)
#pragma unroll
    for (int r = 0; r < 8; ++r) acc[t][r] = 0.f;

  const int steps = K / 32;
  async_stage_a16(A, K, row0, 0, lds_a[0]);
  for (int i = 0; i < steps; ++i) {
    if (i + 1 < steps) {
      async_stage_a16(A, K, row0, (i + 1) * 32, lds_a[(i + 1) & 1]);
      wait_async_le1();
    } else {
      wait_async_le0();
    }
    __syncthreads();
    v16bf a = lds_frag_a(lds_a[i & 1]);
#pragma unroll
    for (int t = 0; t < 4; ++t)
      acc[t] = wmma_bf16(a, load_frag_b(W, K, n0 + t * 16, i * 32), acc[t]);
    __syncthreads();
  }

#pragma unroll
  for (int t = 0; t < 4; ++t) {
    const int col = n0 + t * 16 + ln;
    const float bb = bias[col];
#pragma unroll
    for (int r = 0; r < 8; ++r)
      tile[(r + g * 8) * DMODEL + col] = acc[t][r] + bb;
  }
  __syncthreads();

  // LayerNorm: 16 threads per row (lanes of a half-group), 32 cols each.
  const float gm = *gamma, bt = *beta;
  const int row = wave * 2 + g;                  // 0..15
  const float* trow = &tile[row * DMODEL];
  float sum = 0.f, sq = 0.f;
#pragma unroll
  for (int i = 0; i < 32; ++i) {
    const float v = trow[ln + i * 16];
    sum += v; sq += v * v;
  }
#pragma unroll
  for (int off = 1; off <= 8; off <<= 1) {
    sum += __shfl_xor(sum, off, 32);
    sq  += __shfl_xor(sq,  off, 32);
  }
  const float mean = sum * (1.f / (float)DMODEL);
  const float var  = (sq - (float)DMODEL * mean * mean) * (1.f / (float)(DMODEL - 1));
  const float rstd = rsqrtf(var + 1e-6f);

  const size_t rbase = (size_t)(row0 + row) * DMODEL;
#pragma unroll
  for (int i = 0; i < 32; ++i) {
    const int c = ln + i * 16;
    const float v = trow[c];
    const float o = res[rbase + c] + gm * (v - mean) * rstd + bt;
    out_f32[rbase + c] = o;
    if (out_bf) out_bf[rbase + c] = f2bf(o);
  }
}

// ---------------- host side ----------------
extern "C" void kernel_launch(void* const* d_in, const int* in_sizes, int n_in,
                              void* d_out, int out_size, void* d_ws, size_t ws_size,
                              hipStream_t stream) {
  (void)in_sizes; (void)n_in; (void)out_size; (void)ws_size;
  const float* x  = (const float*)d_in[0];
  const unsigned char* mask = (const unsigned char*)d_in[1];   // jnp.bool_ = 1 byte
  const float* wq = (const float*)d_in[2];  const float* bq = (const float*)d_in[3];
  const float* wk = (const float*)d_in[4];  const float* bk = (const float*)d_in[5];
  const float* wv = (const float*)d_in[6];  const float* bv = (const float*)d_in[7];
  const float* wo = (const float*)d_in[8];  const float* bo = (const float*)d_in[9];
  const float* w1 = (const float*)d_in[10]; const float* b1 = (const float*)d_in[11];
  const float* w2 = (const float*)d_in[12]; const float* b2 = (const float*)d_in[13];
  const float* g1 = (const float*)d_in[14]; const float* beta1 = (const float*)d_in[15];
  const float* g2 = (const float*)d_in[16]; const float* beta2 = (const float*)d_in[17];
  float* out = (float*)d_out;

  char* ws = (char*)d_ws;
  auto alloc = [&](size_t bytes) {
    char* p = ws;
    ws += (bytes + 255) & ~(size_t)255;
    return p;
  };
  bf16_t* xb  = (bf16_t*)alloc((size_t)BSROWS * DMODEL * 2);
  bf16_t* wqb = (bf16_t*)alloc((size_t)DMODEL * DMODEL * 2);
  bf16_t* wkb = (bf16_t*)alloc((size_t)DMODEL * DMODEL * 2);
  bf16_t* wvb = (bf16_t*)alloc((size_t)DMODEL * DMODEL * 2);
  bf16_t* wob = (bf16_t*)alloc((size_t)DMODEL * DMODEL * 2);
  bf16_t* w1b = (bf16_t*)alloc((size_t)DFFN * DMODEL * 2);
  bf16_t* w2b = (bf16_t*)alloc((size_t)DMODEL * DFFN * 2);
  bf16_t* qbuf = (bf16_t*)alloc((size_t)BSROWS * DMODEL * 2);   // [B,H,S,DK]
  bf16_t* kbuf = (bf16_t*)alloc((size_t)BSROWS * DMODEL * 2);   // [B,H,S,DK]
  bf16_t* vTbuf = (bf16_t*)alloc((size_t)BSROWS * DMODEL * 2);  // [B,H,DK,S]
  bf16_t* obuf = (bf16_t*)alloc((size_t)BSROWS * DMODEL * 2);   // [BS, D]
  float*  ybuf = (float*)alloc((size_t)BSROWS * DMODEL * 4);
  bf16_t* ybb  = (bf16_t*)alloc((size_t)BSROWS * DMODEL * 2);
  bf16_t* hbuf = (bf16_t*)alloc((size_t)BSROWS * DFFN * 2);

  auto cvt = [&](const float* src, bf16_t* dst, size_t n) {
    int n4 = (int)(n / 4);
    cvt_f32_bf16<<<(n4 + 255) / 256, 256, 0, stream>>>(src, dst, n4);
  };
  cvt(x,  xb,  (size_t)BSROWS * DMODEL);
  cvt(wq, wqb, (size_t)DMODEL * DMODEL);
  cvt(wk, wkb, (size_t)DMODEL * DMODEL);
  cvt(wv, wvb, (size_t)DMODEL * DMODEL);
  cvt(wo, wob, (size_t)DMODEL * DMODEL);
  cvt(w1, w1b, (size_t)DFFN * DMODEL);
  cvt(w2, w2b, (size_t)DMODEL * DFFN);

  // QKV projections
  gemm_qkv<<<dim3(BSROWS / 128, DMODEL / 64, 3), 256, 0, stream>>>(
      xb, wqb, wkb, wvb, bq, bk, bv, qbuf, kbuf, vTbuf);

  // flash attention
  flash_attn<<<dim3(BATCH * NHEAD, SEQ / 64), 128, 0, stream>>>(
      qbuf, kbuf, vTbuf, mask, obuf);

  // O-projection + LayerNorm + residual -> y (fp32 + bf16)
  gemm_ln_res<<<BSROWS / 16, 256, 0, stream>>>(
      obuf, wob, bo, x, g1, beta1, ybuf, ybb, DMODEL);

  // FFN layer 1 (+ReLU)
  gemm_ffn1<<<dim3(BSROWS / 128, DFFN / 64), 256, 0, stream>>>(
      ybb, w1b, b1, hbuf);

  // FFN layer 2 + LayerNorm + residual -> final output
  gemm_ln_res<<<BSROWS / 16, 256, 0, stream>>>(
      hbuf, w2b, b2, ybuf, g2, beta2, out, nullptr, DFFN);
}
